// SS2D_44985487458721
// MI455X (gfx1250) — matrix operand order repaired and checked by reading, not compiled, not run on hardware
//
#include <hip/hip_runtime.h>
#include <hip/hip_bf16.h>

#define NB    4       // batch
#define HH_   64
#define WW_   64
#define LL    4096    // H*W
#define DM    96      // d_model
#define DI    192     // d_inner
#define NS    16      // n_state
#define RR    6       // dt rank
#define KD    4       // directions
#define CPAD  48      // 38 rows padded to 48 (3 WMMA n-tiles)
#define CROWS 38      // R + 2N

#define USE_TDM 1     // stage scan chunks with tensor_load_to_lds (TDM)

typedef __attribute__((ext_vector_type(16))) __bf16 bf16x16;
typedef __attribute__((ext_vector_type(8)))  __bf16 bf16x8;
typedef __attribute__((ext_vector_type(8)))  float  f32x8;
typedef __attribute__((ext_vector_type(4)))  unsigned int u32x4;
typedef __attribute__((ext_vector_type(8)))  int  i32x8;
typedef __attribute__((ext_vector_type(4)))  int  i32x4;

__device__ __forceinline__ __bf16 f2bf(float f) {
    union { float f; unsigned u; } in; in.f = f;
    unsigned u = in.u;
    unsigned r = u + 0x7FFFu + ((u >> 16) & 1u);   // round-to-nearest-even
    union { unsigned short s; __bf16 b; } out;
    out.s = (unsigned short)(r >> 16);
    return out.b;
}

// A-operand per-lane layout (16-bit 16x32): halves 0..7 = K[k0+8*hi .. +8],
// halves 8..15 = K[k0+16+8*hi .. +8]
__device__ __forceinline__ bf16x16 loadA16(const __bf16* p) {
    union { bf16x16 v; bf16x8 h[2]; } u;
    u.h[0] = *(const bf16x8*)(p);
    u.h[1] = *(const bf16x8*)(p + 16);
    return u.v;
}
// B-operand per-lane layout (16-bit 32x16): halves 0..7 = K[k0+8*hi .. +8],
// halves 8..15 = K[k0+16+8*hi .. +8]
__device__ __forceinline__ bf16x16 loadB16(const __bf16* p) {
    union { bf16x16 v; bf16x8 h[2]; } u;
    u.h[0] = *(const bf16x8*)(p);
    u.h[1] = *(const bf16x8*)(p + 16);
    return u.v;
}

__device__ __forceinline__ float sigmoidf_(float x) { return 1.0f / (1.0f + __expf(-x)); }
__device__ __forceinline__ float softplusf_(float x) {
    return (x > 20.0f) ? x : log1pf(__expf(x));
}

// ---------------- conversion kernels ----------------
__global__ void to_bf16_kernel(const float* __restrict__ src, __bf16* __restrict__ dst, long n) {
    long i = (long)blockIdx.x * blockDim.x + threadIdx.x;
    if (i < n) dst[i] = f2bf(src[i]);
}

// x_proj_weight (K,38,192) -> (K,48,192) bf16, rows 38..47 zero
__global__ void pad_xproj_kernel(const float* __restrict__ src, __bf16* __restrict__ dst) {
    int i = blockIdx.x * blockDim.x + threadIdx.x;
    if (i >= KD * CPAD * DI) return;
    int d = i % DI;
    int c = (i / DI) % CPAD;
    int k = i / (DI * CPAD);
    float v = (c < CROWS) ? src[((long)k * CROWS + c) * DI + d] : 0.0f;
    dst[i] = f2bf(v);
}

// ---------------- WMMA GEMM:  C[m,n] = sum_k A[m,k] * W[n,k]  ----------------
// A: (M,K) bf16 row-major, W: (N,K) bf16 row-major. 4 waves/block, 16 rows/wave,
// NT 16-col tiles per wave. If TC, store C transposed: C[n*ldcT + m].
template<int NT, bool TC>
__global__ __launch_bounds__(128)
void gemm_bf16_kernel(const __bf16* __restrict__ A, const __bf16* __restrict__ W,
                      float* __restrict__ C, int M, int N, int K,
                      long strideA, long strideW, int wmod, long strideC, int ldcT) {
    const int lane = threadIdx.x & 31;
    const int wave = threadIdx.x >> 5;
    const int m0 = (blockIdx.x * 4 + wave) * 16;
    const int n0 = blockIdx.y * (NT * 16);
    const int z  = blockIdx.z;
    const __bf16* Ab = A + (long)z * strideA;
    const __bf16* Wb = W + (long)(z % wmod) * strideW;
    float* Cb = C + (long)z * strideC;
    const int lm = lane & 15;
    const int hi = lane >> 4;

    f32x8 acc[NT];
#pragma unroll
    for (int j = 0; j < NT; j++)
#pragma unroll
        for (int e = 0; e < 8; e++) acc[j][e] = 0.0f;

    const __bf16* arow = Ab + (long)(m0 + lm) * K;
    for (int k0 = 0; k0 < K; k0 += 32) {
        bf16x16 a = loadA16(arow + k0 + hi * 8);
#pragma unroll
        for (int j = 0; j < NT; j++) {
            const __bf16* wrow = Wb + (long)(n0 + j * 16 + lm) * K;
            bf16x16 b = loadB16(wrow + k0 + hi * 8);
            acc[j] = __builtin_amdgcn_wmma_f32_16x16x32_bf16(
                false, a, false, b, (short)0, acc[j], false, false);
        }
    }
#pragma unroll
    for (int j = 0; j < NT; j++) {
        int n = n0 + j * 16 + lm;
#pragma unroll
        for (int r = 0; r < 8; r++) {
            int m = m0 + r + hi * 8;
            if (TC) Cb[(long)n * ldcT + m] = acc[j][r];
            else    Cb[(long)m * N + n]    = acc[j][r];
        }
    }
}

// ---------------- depthwise 3x3 conv + SiLU ----------------
// xz: (B, L, 384) fp32 (cols 0..191 = xc). out xc_conv: (B, L, 192)
__global__ void conv_silu_kernel(const float* __restrict__ xz, const float* __restrict__ cw,
                                 const float* __restrict__ cb, float* __restrict__ xc) {
    long idx = (long)blockIdx.x * blockDim.x + threadIdx.x;
    if (idx >= (long)NB * LL * DI) return;
    int d  = (int)(idx % DI);
    long t = idx / DI;
    int pp = (int)(t & (LL - 1));
    int b  = (int)(t >> 12);
    int h  = pp >> 6, w = pp & 63;
    float acc = cb[d];
#pragma unroll
    for (int ky = 0; ky < 3; ky++) {
        int y = h + ky - 1;
        if (y < 0 || y >= HH_) continue;
#pragma unroll
        for (int kx = 0; kx < 3; kx++) {
            int x = w + kx - 1;
            if (x < 0 || x >= WW_) continue;
            float v = xz[((long)b * LL + y * WW_ + x) * (2 * DI) + d];
            acc = fmaf(v, cw[d * 9 + ky * 3 + kx], acc);
        }
    }
    xc[idx] = acc * sigmoidf_(acc);
}

// pixel index visited at step l of direction k
__device__ __forceinline__ int dir_pix(int k, int l) {
    int li = (k & 2) ? (LL - 1 - l) : l;
    if (k & 1) { int h = li & 63; int w = li >> 6; return h * WW_ + w; }
    return li;
}

// ---------------- build xs (B,K,L,192) bf16 from xc ----------------
__global__ void build_xs_kernel(const float* __restrict__ xc, __bf16* __restrict__ xs) {
    long idx = (long)blockIdx.x * blockDim.x + threadIdx.x;
    if (idx >= (long)NB * KD * LL * DI) return;
    int d  = (int)(idx % DI);
    long t = idx / DI;
    int l  = (int)(t & (LL - 1));
    int k  = (int)((t >> 12) & 3);
    int b  = (int)(t >> 14);
    int p  = dir_pix(k, l);
    xs[idx] = f2bf(xc[((long)b * LL + p) * DI + d]);
}

// ---------------- selective scan ----------------
// one block per (b,k); thread d holds h[0..15] in registers.
// xdbl: (B*K, 48, L) fp32 [rows 0..5 dt-rank, 6..21 B, 22..37 C]
// ys out: (B*K, L, 192) fp32
#define CHUNK 64
__global__ __launch_bounds__(192)
void scan_kernel(const float* __restrict__ xdbl, const float* __restrict__ xc,
                 const float* __restrict__ dtw, const float* __restrict__ dtb,
                 const float* __restrict__ Alogs, const float* __restrict__ Ds,
                 float* __restrict__ ys) {
    const int z = blockIdx.x;          // b*4 + k
    const int b = z >> 2, k = z & 3;
    const int d = threadIdx.x;

    __shared__ float sbuf[CROWS * CHUNK];

    float w6[RR];
#pragma unroll
    for (int r = 0; r < RR; r++) w6[r] = dtw[((long)k * DI + d) * RR + r];
    const float bias = dtb[k * DI + d];
    const float Dv   = Ds[k * DI + d];
    float an[NS];
#pragma unroll
    for (int n = 0; n < NS; n++) an[n] = -__expf(Alogs[((long)(k * DI + d)) * NS + n]);
    float h[NS];
#pragma unroll
    for (int n = 0; n < NS; n++) h[n] = 0.0f;

    const float* xdb = xdbl + (long)z * CPAD * LL;

#if USE_TDM
    const unsigned lds_a = (unsigned)(unsigned long long)(uintptr_t)(void*)sbuf;
#endif

    for (int l0 = 0; l0 < LL; l0 += CHUNK) {
        __syncthreads();                       // previous chunk fully consumed
#if USE_TDM
        // TDM: DMA a 2D tile (38 rows x 64 fp32, row stride 4096 elems) into LDS.
        if (threadIdx.x < 32) {                // one wave issues; TENSORcnt is per-wave
            unsigned long long ga = (unsigned long long)(uintptr_t)(xdb + l0);
            u32x4 g0;
            g0[0] = 1u;                                        // count=1 (valid), user mode
            g0[1] = lds_a;                                     // LDS byte address
            g0[2] = (unsigned)(ga & 0xFFFFFFFFu);              // global_addr[31:0]
            g0[3] = (unsigned)((ga >> 32) & 0x01FFFFFFu)       // global_addr[56:32]
                  | (2u << 30);                                // type=2 ("image")
            i32x8 g1;
            g1[0] = (int)(2u << 16);                           // data_size=4B; no mask/pad
            g1[1] = 0;                                         // td0[15:0]=0 (td0=2^30)
            g1[2] = 0x4000;                                    // td0[31:16]; td1[15:0]=0
            g1[3] = (int)((64u << 16) | 0x4000u);              // tile_dim0=64; td1[31:16]
            g1[4] = CROWS;                                     // tile_dim1=38; tile_dim2=0
            g1[5] = LL;                                        // tensor_dim0_stride=4096
            g1[6] = 0;                                         // stride0 hi, stride1 lo
            g1[7] = 0;                                         // stride1 hi
            i32x4 gz4;
            gz4[0] = 0; gz4[1] = 0; gz4[2] = 0; gz4[3] = 0;    // 2D tensor: groups 2/3 unused
            i32x8 gz8;
#pragma unroll
            for (int e = 0; e < 8; e++) gz8[e] = 0;            // trailing group (6-arg form)
            __builtin_amdgcn_tensor_load_to_lds(g0, g1, gz4, gz4, gz8, 0);
            __builtin_amdgcn_s_wait_tensorcnt((short)0);
        }
        __syncthreads();                       // LDS tile visible to all 6 waves
#else
        for (int i = threadIdx.x; i < CROWS * CHUNK; i += DI) {
            int c = i / CHUNK, l = i % CHUNK;
            sbuf[i] = xdb[(long)c * LL + l0 + l];
        }
        __syncthreads();
#endif
        if (l0 + CHUNK < LL)
            __builtin_prefetch(xdb + (long)(l0 + CHUNK), 0, 1);

        for (int l = 0; l < CHUNK; l++) {
            const int lg = l0 + l;
            const int p  = dir_pix(k, lg);
            const float u = xc[((long)b * LL + p) * DI + d];
            float dt = bias;
#pragma unroll
            for (int r = 0; r < RR; r++) dt = fmaf(w6[r], sbuf[r * CHUNK + l], dt);
            dt = softplusf_(dt);
            const float du = dt * u;
            float y = 0.0f;
#pragma unroll
            for (int n = 0; n < NS; n++) {
                float Bn = sbuf[(RR + n) * CHUNK + l];
                float Cn = sbuf[(RR + NS + n) * CHUNK + l];
                h[n] = fmaf(h[n], __expf(dt * an[n]), du * Bn);
                y    = fmaf(h[n], Cn, y);
            }
            y = fmaf(u, Dv, y);
            ys[((long)z * LL + lg) * DI + d] = y;
        }
    }
}

// ---------------- combine 4 directions + LayerNorm + z-gate -> bf16 ----------------
__global__ __launch_bounds__(192)
void combine_ln_kernel(const float* __restrict__ ys, const float* __restrict__ xz,
                       const float* __restrict__ nw, const float* __restrict__ nb,
                       __bf16* __restrict__ ybf) {
    const int bp = blockIdx.x;              // b*L + p
    const int b = bp >> 12, pp = bp & (LL - 1);
    const int d = threadIdx.x;
    const int h = pp >> 6, w = pp & 63;
    const int l1 = w * HH_ + h;             // step index of dir1 visiting this pixel

    float y = ys[(((long)(b * 4 + 0) * LL + pp)           ) * DI + d]
            + ys[(((long)(b * 4 + 1) * LL + l1)           ) * DI + d]
            + ys[(((long)(b * 4 + 2) * LL + (LL - 1 - pp))) * DI + d]
            + ys[(((long)(b * 4 + 3) * LL + (LL - 1 - l1))) * DI + d];

    __shared__ float s[DI];
    // mean
    s[d] = y; __syncthreads();
    if (d < 96) s[d] += s[d + 96]; __syncthreads();
    if (d < 48) s[d] += s[d + 48]; __syncthreads();
    if (d < 24) s[d] += s[d + 24]; __syncthreads();
    if (d < 12) s[d] += s[d + 12]; __syncthreads();
    if (d < 6)  s[d] += s[d + 6];  __syncthreads();
    if (d == 0) { float t = 0; for (int i = 0; i < 6; i++) t += s[i]; s[0] = t; }
    __syncthreads();
    const float mu = s[0] * (1.0f / DI);
    __syncthreads();
    // variance
    float c = y - mu;
    s[d] = c * c; __syncthreads();
    if (d < 96) s[d] += s[d + 96]; __syncthreads();
    if (d < 48) s[d] += s[d + 48]; __syncthreads();
    if (d < 24) s[d] += s[d + 24]; __syncthreads();
    if (d < 12) s[d] += s[d + 12]; __syncthreads();
    if (d < 6)  s[d] += s[d + 6];  __syncthreads();
    if (d == 0) { float t = 0; for (int i = 0; i < 6; i++) t += s[i]; s[0] = t; }
    __syncthreads();
    const float var = s[0] * (1.0f / DI);

    float yn = nw[d] * c * rsqrtf(var + 1e-5f) + nb[d];
    float zv = xz[((long)b * LL + pp) * (2 * DI) + DI + d];
    ybf[(long)bp * DI + d] = f2bf(yn * (zv * sigmoidf_(zv)));
}

// ---------------- workspace layout (bytes) ----------------
#define OFF_XZ    0UL                               // (B,L,384) f32   25,165,824
#define OFF_XC    25165824UL                        // (B,L,192) f32   12,582,912
#define OFF_XSBF  37748736UL                        // (B,K,L,192) bf16 25,165,824
#define OFF_XDBL  62914560UL                        // (B*K,48,L) f32  12,582,912
#define OFF_YS    75497472UL                        // (B*K,L,192) f32 50,331,648
#define OFF_YBF   125829120UL                       // (B,L,192) bf16   6,291,456
#define OFF_XBF   132120576UL                       // (B,L,96) bf16    3,145,728
#define OFF_WI    135266304UL                       // (384,96) bf16       73,728
#define OFF_WX    135340032UL                       // (4,48,192) bf16     73,728
#define OFF_WO    135413760UL                       // (96,192) bf16       36,864

extern "C" void kernel_launch(void* const* d_in, const int* in_sizes, int n_in,
                              void* d_out, int out_size, void* d_ws, size_t ws_size,
                              hipStream_t stream) {
    (void)in_sizes; (void)n_in; (void)out_size; (void)ws_size;
    const float* x       = (const float*)d_in[0];
    const float* inw     = (const float*)d_in[1];
    const float* convw   = (const float*)d_in[2];
    const float* convb   = (const float*)d_in[3];
    const float* xprojw  = (const float*)d_in[4];
    const float* dtw     = (const float*)d_in[5];
    const float* dtb     = (const float*)d_in[6];
    const float* Alogs   = (const float*)d_in[7];
    const float* Ds      = (const float*)d_in[8];
    const float* nw      = (const float*)d_in[9];
    const float* nb      = (const float*)d_in[10];
    const float* outw    = (const float*)d_in[11];

    char* ws = (char*)d_ws;
    float*  xz    = (float*) (ws + OFF_XZ);
    float*  xc    = (float*) (ws + OFF_XC);
    __bf16* xsbf  = (__bf16*)(ws + OFF_XSBF);
    float*  xdbl  = (float*) (ws + OFF_XDBL);
    float*  ys    = (float*) (ws + OFF_YS);
    __bf16* ybf   = (__bf16*)(ws + OFF_YBF);
    __bf16* xbf   = (__bf16*)(ws + OFF_XBF);
    __bf16* wibf  = (__bf16*)(ws + OFF_WI);
    __bf16* wxbf  = (__bf16*)(ws + OFF_WX);
    __bf16* wobf  = (__bf16*)(ws + OFF_WO);

    const long M = (long)NB * LL;            // 16384 rows

    // 1) fp32 -> bf16 conversions
    to_bf16_kernel<<<(unsigned)((M * DM + 255) / 256), 256, 0, stream>>>(x, xbf, M * DM);
    to_bf16_kernel<<<(2 * DI * DM + 255) / 256, 256, 0, stream>>>(inw, wibf, (long)2 * DI * DM);
    to_bf16_kernel<<<(DM * DI + 255) / 256, 256, 0, stream>>>(outw, wobf, (long)DM * DI);
    pad_xproj_kernel<<<(KD * CPAD * DI + 255) / 256, 256, 0, stream>>>(xprojw, wxbf);

    // 2) in_proj GEMM: (16384x96)x(384x96)^T -> xz (16384x384)
    gemm_bf16_kernel<4, false><<<dim3(256, 6, 1), 128, 0, stream>>>(
        xbf, wibf, xz, (int)M, 2 * DI, DM, 0L, 0L, 1, 0L, 0);

    // 3) depthwise conv 3x3 + SiLU -> xc (B,L,192)
    conv_silu_kernel<<<(unsigned)((M * DI + 255) / 256), 256, 0, stream>>>(xz, convw, convb, xc);

    // 4) gather 4 scan directions -> xs bf16 (B,K,L,192)
    build_xs_kernel<<<(unsigned)((M * KD * DI + 255) / 256), 256, 0, stream>>>(xc, xsbf);

    // 5) x_proj GEMM per (b,k): (4096x192)x(48x192)^T, stored transposed (48,4096)
    gemm_bf16_kernel<3, true><<<dim3(64, 1, NB * KD), 128, 0, stream>>>(
        xsbf, wxbf, xdbl, LL, CPAD, DI,
        (long)LL * DI, (long)CPAD * DI, KD, (long)CPAD * LL, LL);

    // 6) selective scan (TDM-staged chunks) -> ys (B*K,L,192)
    scan_kernel<<<NB * KD, DI, 0, stream>>>(xdbl, xc, dtw, dtb, Alogs, Ds, ys);

    // 7) combine directions + LayerNorm + z gate -> ybf (B,L,192) bf16
    combine_ln_kernel<<<(unsigned)M, DI, 0, stream>>>(ys, xz, nw, nb, ybf);

    // 8) out_proj GEMM: (16384x192)x(96x192)^T -> d_out (16384x96) fp32
    gemm_bf16_kernel<6, false><<<dim3(256, 1, 1), 128, 0, stream>>>(
        ybf, wobf, (float*)d_out, (int)M, DM, DI, 0L, 0L, 1, 0L, 0);
}
